// TtMoeLayer_79534204387974
// MI455X (gfx1250) — matrix operand order repaired and checked by reading, not compiled, not run on hardware
//
#include <hip/hip_runtime.h>

// ---------------------------------------------------------------------------
// MoE (dense all-experts) for MI455X / gfx1250 — bf16 WMMA, async-to-LDS,
// double-buffered LDS pipeline.
//   E=8, S=128, B=32 -> TOK=4096 tokens, H=1024, F=4096.
// Per call:
//   1) gate_kernel : logits = x@gate_w, softmax, emulated top-2 -> ew[TOK][E]
//   2) cvt_kernel  : x fp32 -> bf16 (xb)
//   3) for e in 0..7:
//        repack w1/w3/w2 : fp32 [K][N] -> bf16 [N][K]   (one-time transpose)
//        gemm1 : h = xb@w1, g = xb@w3, act = silu(h)*g  (bf16)
//        gemm2 : out (+)= ew[:,e] * (act @ w2)          (fp32)
// Workspace: ew@0 (128KB), xb@1MB (8MB), act@16MB (32MB),
//            w1t@48MB, w3t@58MB, w2t@68MB (8.4MB each, reused per expert).
// ---------------------------------------------------------------------------

typedef unsigned short u16;
typedef unsigned int   u32;
typedef unsigned long long u64;
typedef __attribute__((ext_vector_type(16))) __bf16 v16bf;
typedef __attribute__((ext_vector_type(8)))  __bf16 v8bf;
typedef __attribute__((ext_vector_type(8)))  float  v8f;

static constexpr int kE   = 8;
static constexpr int kH   = 1024;
static constexpr int kF   = 4096;
static constexpr int kTOK = 4096;

static constexpr int LST = 72;   // LDS row stride (halfwords): 144B = 9*16B

__device__ __forceinline__ u16 f32_to_bf16(float f) {
    u32 u = __float_as_uint(f);
    u32 r = (u + 0x7FFFu + ((u >> 16) & 1u)) >> 16;   // round-to-nearest-even
    return (u16)r;
}

// 16B global -> LDS async copy. ISA §08: IOFFSET is added to BOTH the LDS
// address (VDST vgpr) and the global address, so one vgpr pair + immediate
// offsets covers a contiguous run.
template <int OFF>
__device__ __forceinline__ void async_cp16(u32 lds_byte_off, const u16* gptr) {
    asm volatile("global_load_async_to_lds_b128 %0, %1, off offset:%2"
                 :: "v"(lds_byte_off), "v"((u64)gptr), "n"(OFF) : "memory");
}
__device__ __forceinline__ void async_wait0() {
    asm volatile("s_wait_asynccnt 0" ::: "memory");
}
__device__ __forceinline__ u32 lds_off(const void* p) {
    return (u32)(size_t)p;   // low 32 bits of the pointer give the LDS byte offset (ISA 10.2 aperture rules)
}

// ---------------------------------------------------------------------------
// Gate: one wave per token. logits -> stable softmax -> emulated top-2 -> ew.
// ---------------------------------------------------------------------------
__global__ __launch_bounds__(128) void gate_kernel(
    const float* __restrict__ x, const float* __restrict__ gw,
    float* __restrict__ ew)
{
    const int l  = threadIdx.x & 31;
    const int wv = threadIdx.x >> 5;
    const int t  = blockIdx.x * 4 + wv;
    const float* xr = x + (size_t)t * kH;

    float p[kE];
#pragma unroll
    for (int e = 0; e < kE; ++e) p[e] = 0.f;
    for (int j = 0; j < kH / 32; ++j) {
        float xv = xr[l + 32 * j];
        const float* g = gw + (size_t)(l + 32 * j) * kE;
#pragma unroll
        for (int e = 0; e < kE; ++e) p[e] = fmaf(xv, g[e], p[e]);
    }
#pragma unroll
    for (int e = 0; e < kE; ++e)
        for (int off = 16; off > 0; off >>= 1)
            p[e] += __shfl_xor(p[e], off, 32);

    if (l == 0) {
        float mx = p[0];
#pragma unroll
        for (int e = 1; e < kE; ++e) mx = fmaxf(mx, p[e]);
        float pr[kE], s = 0.f;
#pragma unroll
        for (int e = 0; e < kE; ++e) { pr[e] = __expf(p[e] - mx); s += pr[e]; }
        float inv = 1.f / s;
#pragma unroll
        for (int e = 0; e < kE; ++e) pr[e] *= inv;
        float w0 = pr[0];
#pragma unroll
        for (int e = 1; e < kE; ++e) w0 = fmaxf(w0, pr[e]);
        float w1v = 0.f;
#pragma unroll
        for (int e = 0; e < kE; ++e) {
            float msk = (pr[e] == w0) ? 0.f : pr[e];
            w1v = fmaxf(w1v, msk);
        }
        float renorm = 1.f / (w0 + w1v);
#pragma unroll
        for (int e = 0; e < kE; ++e) {
            float msk = (pr[e] == w0) ? 0.f : pr[e];
            float r = 0.f;
            if (pr[e] == w0) r += renorm * w0;
            if (msk == w1v)  r += renorm * w1v;
            ew[(size_t)t * kE + e] = r;
        }
    }
}

// ---------------------------------------------------------------------------
// fp32 -> bf16 convert (x), 4-wide
// ---------------------------------------------------------------------------
struct alignas(8) us4 { u16 x, y, z, w; };

__global__ __launch_bounds__(256) void cvt_kernel(
    const float4* __restrict__ in, us4* __restrict__ outb, int n4)
{
    int i = blockIdx.x * blockDim.x + threadIdx.x;
    if (i >= n4) return;
    float4 v = in[i];
    us4 r{f32_to_bf16(v.x), f32_to_bf16(v.y), f32_to_bf16(v.z), f32_to_bf16(v.w)};
    outb[i] = r;
}

// ---------------------------------------------------------------------------
// Repack: W fp32 [K][N] -> Wt bf16 [N][K]   (LDS-tiled 64x64 transpose)
// ---------------------------------------------------------------------------
__global__ __launch_bounds__(256) void repack_kernel(
    const float* __restrict__ W, u16* __restrict__ Wt, int K, int N)
{
    __shared__ u16 tile[64 * 66];
    const int kb = blockIdx.x * 64, nb = blockIdx.y * 64;
    const int t = threadIdx.x;
#pragma unroll
    for (int i = 0; i < 16; ++i) {
        int idx = t + i * 256;
        int k = idx >> 6, n = idx & 63;
        tile[n * 66 + k] = f32_to_bf16(W[(size_t)(kb + k) * N + nb + n]);
    }
    __syncthreads();
#pragma unroll
    for (int i = 0; i < 16; ++i) {
        int idx = t + i * 256;
        int n = idx >> 6, k = idx & 63;
        Wt[(size_t)(nb + n) * K + kb + k] = tile[n * 66 + k];
    }
}

// ---------------------------------------------------------------------------
// WMMA fragment helpers (16-bit A 16x32 / B 32x16 layouts, ISA 7.12.2)
// ---------------------------------------------------------------------------
__device__ __forceinline__ v16bf load_a_frag(const u16* sA, int row, int kc, int hl) {
    const u16* p = sA + row * LST + kc * 32 + hl * 16;
    v8bf lo = *(const v8bf*)p;          // K = base .. base+7
    v8bf hi = *(const v8bf*)(p + 8);    // K = base+8 .. base+15
    v16bf a;
#pragma unroll
    for (int i = 0; i < 8; ++i) { a[i] = lo[i]; a[i + 8] = hi[i]; }
    return a;
}
__device__ __forceinline__ v16bf load_b_frag(const u16* sB, int nrow, int kc, int hl) {
    return *(const v16bf*)(sB + nrow * LST + kc * 32 + hl * 16);
}

// ---------------------------------------------------------------------------
// GEMM1: xb bf16 [TOK][kH] x {w1t,w3t} bf16 [kF][kH] -> act = silu(h)*g
// Block 128x64, 8 waves (wave tile 32x32), K-step 64, double-buffered LDS.
// ---------------------------------------------------------------------------
__global__ __launch_bounds__(256) void gemm1_kernel(
    const u16* __restrict__ A,     // xb  [TOK][kH]
    const u16* __restrict__ B1,    // w1t [kF][kH]
    const u16* __restrict__ B3,    // w3t [kF][kH]
    u16*       __restrict__ act)   // [TOK][kF]
{
    __shared__ u16 sA [2][128 * LST];
    __shared__ u16 sB1[2][ 64 * LST];
    __shared__ u16 sB3[2][ 64 * LST];

    const int t  = threadIdx.x;
    const int l  = t & 31, wv = t >> 5;
    const int wm = wv & 3, wn = wv >> 2;
    const int hl = l >> 4, ln = l & 15;
    const int Mb = blockIdx.y * 128;
    const int Nb = blockIdx.x * 64;

    const int ar = t >> 1, ah = t & 1;     // A: 128 rows x 128B, 64B/thread
    const int br = t >> 2, bq = t & 3;     // B:  64 rows x 128B, 32B/thread

    const u16* gA  = A  + (size_t)(Mb + ar) * kH + ah * 32;
    const u16* gB1 = B1 + (size_t)(Nb + br) * kH + bq * 16;
    const u16* gB3 = B3 + (size_t)(Nb + br) * kH + bq * 16;

    auto stage = [&](int buf, int k0) {
        u32 dA = lds_off(&sA[buf][ar * LST + ah * 32]);
        const u16* pA = gA + k0;
        async_cp16<0>(dA, pA);  async_cp16<16>(dA, pA);
        async_cp16<32>(dA, pA); async_cp16<48>(dA, pA);
        u32 d1 = lds_off(&sB1[buf][br * LST + bq * 16]);
        u32 d3 = lds_off(&sB3[buf][br * LST + bq * 16]);
        async_cp16<0>(d1, gB1 + k0); async_cp16<16>(d1, gB1 + k0);
        async_cp16<0>(d3, gB3 + k0); async_cp16<16>(d3, gB3 + k0);
    };

    v8f accH[2][2] = {};
    v8f accG[2][2] = {};

    stage(0, 0);
    async_wait0();
    __syncthreads();

    int p = 0;
    for (int k0 = 0; k0 < kH; k0 += 64) {
        if (k0 + 64 < kH) stage(p ^ 1, k0 + 64);   // overlap DMA with WMMA

#pragma unroll
        for (int kc = 0; kc < 2; ++kc) {
            v16bf a[2], b1[2], b3[2];
#pragma unroll
            for (int mt = 0; mt < 2; ++mt)
                a[mt] = load_a_frag(sA[p], wm * 32 + mt * 16 + ln, kc, hl);
#pragma unroll
            for (int nt = 0; nt < 2; ++nt) {
                b1[nt] = load_b_frag(sB1[p], wn * 32 + nt * 16 + ln, kc, hl);
                b3[nt] = load_b_frag(sB3[p], wn * 32 + nt * 16 + ln, kc, hl);
            }
#pragma unroll
            for (int mt = 0; mt < 2; ++mt)
#pragma unroll
                for (int nt = 0; nt < 2; ++nt) {
                    accH[mt][nt] = __builtin_amdgcn_wmma_f32_16x16x32_bf16(
                        false, a[mt], false, b1[nt], (short)0, accH[mt][nt], false, false);
                    accG[mt][nt] = __builtin_amdgcn_wmma_f32_16x16x32_bf16(
                        false, a[mt], false, b3[nt], (short)0, accG[mt][nt], false, false);
                }
        }
        async_wait0();
        __syncthreads();
        p ^= 1;
    }

    // SwiGLU epilogue (C layout: M = r + hl*8, N = ln)
#pragma unroll
    for (int mt = 0; mt < 2; ++mt)
#pragma unroll
        for (int nt = 0; nt < 2; ++nt)
#pragma unroll
            for (int r = 0; r < 8; ++r) {
                int row = Mb + wm * 32 + mt * 16 + hl * 8 + r;
                int col = Nb + wn * 32 + nt * 16 + ln;
                float hv = accH[mt][nt][r];
                float gv = accG[mt][nt][r];
                float sv = hv * __builtin_amdgcn_rcpf(1.f + __expf(-hv));
                act[(size_t)row * kF + col] = f32_to_bf16(sv * gv);
            }
}

// ---------------------------------------------------------------------------
// GEMM2: act bf16 [TOK][kF] x w2t bf16 [kH][kF] ; epilogue *ew, store/accum.
// ---------------------------------------------------------------------------
__global__ __launch_bounds__(256) void gemm2_kernel(
    const u16* __restrict__ A,     // act [TOK][kF]
    const u16* __restrict__ B2,    // w2t [kH][kF]
    const float* __restrict__ ew,  // [TOK][E]
    float*     __restrict__ out,   // [TOK][kH]
    int e)
{
    __shared__ u16 sA[2][128 * LST];
    __shared__ u16 sB[2][ 64 * LST];

    const int t  = threadIdx.x;
    const int l  = t & 31, wv = t >> 5;
    const int wm = wv & 3, wn = wv >> 2;
    const int hl = l >> 4, ln = l & 15;
    const int Mb = blockIdx.y * 128;
    const int Nb = blockIdx.x * 64;
    const int ar = t >> 1, ah = t & 1;
    const int br = t >> 2, bq = t & 3;

    const u16* gA = A  + (size_t)(Mb + ar) * kF + ah * 32;
    const u16* gB = B2 + (size_t)(Nb + br) * kF + bq * 16;

    auto stage = [&](int buf, int k0) {
        u32 dA = lds_off(&sA[buf][ar * LST + ah * 32]);
        const u16* pA = gA + k0;
        async_cp16<0>(dA, pA);  async_cp16<16>(dA, pA);
        async_cp16<32>(dA, pA); async_cp16<48>(dA, pA);
        u32 dB = lds_off(&sB[buf][br * LST + bq * 16]);
        async_cp16<0>(dB, gB + k0); async_cp16<16>(dB, gB + k0);
    };

    v8f acc[2][2] = {};

    stage(0, 0);
    async_wait0();
    __syncthreads();

    int p = 0;
    for (int k0 = 0; k0 < kF; k0 += 64) {
        if (k0 + 64 < kF) stage(p ^ 1, k0 + 64);

#pragma unroll
        for (int kc = 0; kc < 2; ++kc) {
            v16bf a[2], b[2];
#pragma unroll
            for (int mt = 0; mt < 2; ++mt)
                a[mt] = load_a_frag(sA[p], wm * 32 + mt * 16 + ln, kc, hl);
#pragma unroll
            for (int nt = 0; nt < 2; ++nt)
                b[nt] = load_b_frag(sB[p], wn * 32 + nt * 16 + ln, kc, hl);
#pragma unroll
            for (int mt = 0; mt < 2; ++mt)
#pragma unroll
                for (int nt = 0; nt < 2; ++nt)
                    acc[mt][nt] = __builtin_amdgcn_wmma_f32_16x16x32_bf16(
                        false, a[mt], false, b[nt], (short)0, acc[mt][nt], false, false);
        }
        async_wait0();
        __syncthreads();
        p ^= 1;
    }

#pragma unroll
    for (int mt = 0; mt < 2; ++mt)
#pragma unroll
        for (int nt = 0; nt < 2; ++nt)
#pragma unroll
            for (int r = 0; r < 8; ++r) {
                int row = Mb + wm * 32 + mt * 16 + hl * 8 + r;
                int col = Nb + wn * 32 + nt * 16 + ln;
                float v = acc[mt][nt][r] * ew[(size_t)row * kE + e];
                size_t o = (size_t)row * kH + col;
                if (e != 0) v += out[o];
                out[o] = v;
            }
}

// ---------------------------------------------------------------------------
// Host launcher
// ---------------------------------------------------------------------------
extern "C" void kernel_launch(void* const* d_in, const int* in_sizes, int n_in,
                              void* d_out, int out_size, void* d_ws, size_t ws_size,
                              hipStream_t stream) {
    (void)in_sizes; (void)n_in; (void)out_size; (void)ws_size;

    const float* x      = (const float*)d_in[0];
    const float* gate_w = (const float*)d_in[1];
    const float* w1     = (const float*)d_in[2];  // [E][H][F]
    const float* w3     = (const float*)d_in[3];  // [E][H][F]
    const float* w2     = (const float*)d_in[4];  // [E][F][H]
    float*       out    = (float*)d_out;

    char* ws   = (char*)d_ws;
    float* ew  = (float*)ws;
    u16*   xb  = (u16*)(ws + ((size_t)1  << 20));
    u16*   act = (u16*)(ws + ((size_t)16 << 20));
    u16*   w1t = (u16*)(ws + ((size_t)48 << 20));  // [F][H] bf16
    u16*   w3t = (u16*)(ws + ((size_t)58 << 20));  // [F][H] bf16
    u16*   w2t = (u16*)(ws + ((size_t)68 << 20));  // [H][F] bf16

    gate_kernel<<<kTOK / 4, 128, 0, stream>>>(x, gate_w, ew);

    const int n4 = kTOK * kH / 4;
    cvt_kernel<<<(n4 + 255) / 256, 256, 0, stream>>>((const float4*)x, (us4*)xb, n4);

    for (int e = 0; e < kE; ++e) {
        const float* w1e = w1 + (size_t)e * kH * kF;
        const float* w3e = w3 + (size_t)e * kH * kF;
        const float* w2e = w2 + (size_t)e * kF * kH;
        repack_kernel<<<dim3(kH / 64, kF / 64), 256, 0, stream>>>(w1e, w1t, kH, kF);
        repack_kernel<<<dim3(kH / 64, kF / 64), 256, 0, stream>>>(w3e, w3t, kH, kF);
        repack_kernel<<<dim3(kF / 64, kH / 64), 256, 0, stream>>>(w2e, w2t, kF, kH);
        gemm1_kernel<<<dim3(kF / 64, kTOK / 128), 256, 0, stream>>>(xb, w1t, w3t, act);
        gemm2_kernel<<<dim3(kH / 64, kTOK / 128), 256, 0, stream>>>(act, w2t, ew, out, e);
    }
}
